// EnhancedBIMambaBlock_41506563949104
// MI455X (gfx1250) — matrix operand order repaired and checked by reading, not compiled, not run on hardware
//
#include <hip/hip_runtime.h>
#include <hip/hip_bf16.h>
#include <math.h>

#define BB   8
#define LL   1024
#define DD   256
#define DI   512
#define NS   16
#define KC   4
#define DTRK 16
#define UU   32
#define MT   (BB*LL)   // 8192 tokens

typedef __attribute__((ext_vector_type(16))) __bf16 v16bf;
typedef __attribute__((ext_vector_type(8)))  float  v8f;

__device__ __forceinline__ float sigmoidf_(float x){ return 1.f/(1.f+__expf(-x)); }
__device__ __forceinline__ float softplusf_(float x){ return (x > 20.f) ? x : log1pf(__expf(x)); }

// ---------------------------------------------------------------------------
// Generic GEMM: C[MT,N] = A[MT,K] * W[N,K]^T   (bf16 WMMA, fp32 accumulate)
// KDIM compile-time, multiple of 32 -> no bounds checks, full unroll,
// pure float4 (b128) fragment loads per the ISA 7.12.2 layouts.
// flipA: read A token rows with t -> L-1-t (backward direction input proj)
// mode 1: fused delta epilogue  C = softplus(C + bias[col])
// ---------------------------------------------------------------------------
template<int KDIM>
__global__ void gemm_bf16_wmma(const float* __restrict__ A, int lda, int flipA,
                               const float* __restrict__ W,
                               float* __restrict__ C, int ldc, int Ndim,
                               const float* __restrict__ bias, int mode)
{
    int wave = (blockIdx.x * blockDim.x + threadIdx.x) >> 5;
    int lane = threadIdx.x & 31;
    int ntiles = Ndim >> 4;
    if (wave >= (MT >> 4) * ntiles) return;
    int tileN = (wave % ntiles) << 4;
    int tileM = (wave / ntiles) << 4;
    int half = lane >> 4;
    int m    = lane & 15;

    // A fragment rows: lanes 0-15 and 16-31 both cover rows 0..15 (K split)
    int arow = tileM + m;
    if (flipA) { int b = arow >> 10, t = arow & (LL-1); arow = (b << 10) + (LL-1 - t); }
    const float* Arow = A + (long)arow * lda;
    const float* Wrow = W + (long)(tileN + m) * KDIM;

    v8f acc = {};
#pragma unroll
    for (int k0 = 0; k0 < KDIM; k0 += 32) {
        // A 16x32 bf16 (ISA 7.12.2): per lane elements 0..7  = K k0+8*half ..+7
        //                            elements 8..15 = K k0+16+8*half ..+7
        const float* ap0 = Arow + k0 + 8*half;
        const float* ap1 = ap0 + 16;
        float4 aA = *(const float4*)(ap0);
        float4 aB = *(const float4*)(ap0 + 4);
        float4 aC = *(const float4*)(ap1);
        float4 aD = *(const float4*)(ap1 + 4);
        // B 32x16 bf16: same K order as A: elements 0..7 = K k0+8*half ..+7, 8..15 = K k0+16+8*half ..+7, col = m
        const float* bp = Wrow + k0 + 8*half;
        float4 bA = *(const float4*)(bp);
        float4 bB = *(const float4*)(bp + 4);
        float4 bC = *(const float4*)(bp + 16);
        float4 bD = *(const float4*)(bp + 20);

        v16bf af, bf;
        af[0]=(__bf16)aA.x; af[1]=(__bf16)aA.y; af[2]=(__bf16)aA.z; af[3]=(__bf16)aA.w;
        af[4]=(__bf16)aB.x; af[5]=(__bf16)aB.y; af[6]=(__bf16)aB.z; af[7]=(__bf16)aB.w;
        af[8]=(__bf16)aC.x; af[9]=(__bf16)aC.y; af[10]=(__bf16)aC.z; af[11]=(__bf16)aC.w;
        af[12]=(__bf16)aD.x; af[13]=(__bf16)aD.y; af[14]=(__bf16)aD.z; af[15]=(__bf16)aD.w;
        bf[0]=(__bf16)bA.x; bf[1]=(__bf16)bA.y; bf[2]=(__bf16)bA.z; bf[3]=(__bf16)bA.w;
        bf[4]=(__bf16)bB.x; bf[5]=(__bf16)bB.y; bf[6]=(__bf16)bB.z; bf[7]=(__bf16)bB.w;
        bf[8]=(__bf16)bC.x; bf[9]=(__bf16)bC.y; bf[10]=(__bf16)bC.z; bf[11]=(__bf16)bC.w;
        bf[12]=(__bf16)bD.x; bf[13]=(__bf16)bD.y; bf[14]=(__bf16)bD.z; bf[15]=(__bf16)bD.w;

        acc = __builtin_amdgcn_wmma_f32_16x16x32_bf16(false, af, false, bf,
                                                      (short)0, acc, false, false);
    }
#pragma unroll
    for (int j = 0; j < 8; ++j) {
        int row = tileM + j + 8*half;   // C: VGPR j -> M=j (lanes0-15) / M=j+8
        int col = tileN + m;
        float v = acc[j];
        if (mode == 1) v = softplusf_(v + bias[col]);
        C[(long)row * ldc + col] = v;
    }
}

// ---------------------------------------------------------------------------
// Zero-pad dt_w (DI,16) -> (DI,32) so the delta GEMM has K=32 exactly.
// ---------------------------------------------------------------------------
__global__ void pad_dtw(const float* __restrict__ dtw, float* __restrict__ dtw_pad)
{
    int i = blockIdx.x * blockDim.x + threadIdx.x;
    if (i >= DI * 32) return;
    int k = i & 31, n = i >> 5;
    dtw_pad[i] = (k < DTRK) ? dtw[n * DTRK + k] : 0.f;
}

// ---------------------------------------------------------------------------
// Depthwise causal conv1d (K=4) + SiLU.  xm = xz[:, :, 0:DI]
// ---------------------------------------------------------------------------
__global__ void conv_silu(const float* __restrict__ xz, const float* __restrict__ cw,
                          const float* __restrict__ cb, float* __restrict__ xc)
{
    int i = blockIdx.x * blockDim.x + threadIdx.x;
    if (i >= MT * DI) return;
    int d   = i & (DI-1);
    int tok = i >> 9;
    int t   = tok & (LL-1);
    float a = cb[d];
#pragma unroll
    for (int k = 0; k < KC; ++k) {
        int tt = t - (KC-1) + k;
        if (tt >= 0)
            a = fmaf(cw[d*KC + k], xz[(long)(tok - (KC-1) + k) * (2*DI) + d], a);
    }
    xc[i] = a * sigmoidf_(a);
}

// ---------------------------------------------------------------------------
// Selective SSM scan. One lane per (b,d) channel; h[16] in registers.
// B_t/C_t (shared per (b,t)) staged 16 timesteps at a time through LDS.
// Writes y in place over delta (same-lane read-then-write per element).
// ---------------------------------------------------------------------------
#define DCH 128
__global__ void ssm_scan(const float* __restrict__ xdbl, float* __restrict__ dly,
                         const float* __restrict__ xc, const float* __restrict__ A_log)
{
    __shared__ float BC[16][32];
    int b = blockIdx.x / (DI / DCH);
    int d = (blockIdx.x % (DI / DCH)) * DCH + threadIdx.x;
    float Arow[NS], h[NS];
#pragma unroll
    for (int n = 0; n < NS; ++n) { Arow[n] = -__expf(A_log[d*NS + n]); h[n] = 0.f; }
    long base = (long)b * LL;
    for (int t0 = 0; t0 < LL; t0 += 16) {
        __syncthreads();
        for (int e = threadIdx.x; e < 16*32; e += DCH) {
            int ts = e >> 5, c = e & 31;
            BC[ts][c] = xdbl[(base + t0 + ts) * 48 + DTRK + c];
        }
        __syncthreads();
        for (int ts = 0; ts < 16; ++ts) {
            long idx = (base + t0 + ts) * DI + d;
            float dl = dly[idx];
            float u  = xc[idx];
            float du = dl * u;
            float y  = 0.f;
#pragma unroll
            for (int n = 0; n < NS; ++n) {
                float e = __expf(dl * Arow[n]);
                h[n] = fmaf(e, h[n], BC[ts][n] * du);
                y    = fmaf(h[n], BC[ts][NS + n], y);
            }
            dly[idx] = y;
        }
    }
}

// ---------------------------------------------------------------------------
// y = (ys + xc*Dp) * silu(res)   written in place into xc
// ---------------------------------------------------------------------------
__global__ void mix_gate_res(float* __restrict__ xc, const float* __restrict__ ys,
                             const float* __restrict__ xz, const float* __restrict__ Dp)
{
    int i = blockIdx.x * blockDim.x + threadIdx.x;
    if (i >= MT * DI) return;
    int d   = i & (DI-1);
    int tok = i >> 9;
    float res = xz[(long)tok * (2*DI) + DI + d];
    xc[i] = (ys[i] + xc[i] * Dp[d]) * (res * sigmoidf_(res));
}

// ---------------------------------------------------------------------------
// S[p] = sum_d dpw[p,d]   (dfull broadcasts a scalar across D)
// ---------------------------------------------------------------------------
__global__ void rowsum64(const float* __restrict__ dpw, float* __restrict__ S)
{
    int p = threadIdx.x;
    if (p < 64) {
        float s = 0.f;
        for (int dd = 0; dd < DD; ++dd) s += dpw[p*DD + dd];
        S[p] = s;
    }
}

// proj[tok,p] = tanh(df[tok] * S[p] + pb[p]); df in (possibly flipped) space
__global__ void dir_proj(const float* __restrict__ x, const float* __restrict__ S,
                         const float* __restrict__ pb, float* __restrict__ proj, int flip)
{
    int i = blockIdx.x * blockDim.x + threadIdx.x;
    if (i >= MT * 64) return;
    int p   = i & 63;
    int tok = i >> 6;
    int b   = tok >> 10;
    int t   = tok & (LL-1);
    float df = 0.f;
    if (t < LL-1) {
        int t1 = flip ? (LL-2 - t) : (t + 1);
        int t0 = flip ? (LL-1 - t) : t;
        df = x[((long)(b << 10) + t1) * DD] - x[((long)(b << 10) + t0) * DD];
    }
    proj[i] = tanhf(fmaf(df, S[p], pb[p]));
}

// acc[b, treal, d] += out * (1 + 0.1*sigmoid(dgw[d,:].proj + dgb[d]))
__global__ void gate_accum(const float* __restrict__ outb, const float* __restrict__ proj,
                           const float* __restrict__ gw, const float* __restrict__ gb,
                           float* __restrict__ acc, int flip)
{
    int i = blockIdx.x * blockDim.x + threadIdx.x;
    if (i >= MT * DD) return;
    int d   = i & (DD-1);
    int tok = i >> 8;
    int b   = tok >> 10;
    int tp  = tok & (LL-1);
    float g = gb[d];
    const float* pr = proj + (long)tok * 64;
    const float* w  = gw   + (long)d   * 64;
#pragma unroll 8
    for (int p = 0; p < 64; ++p) g = fmaf(w[p], pr[p], g);
    g = sigmoidf_(g);
    int tr = flip ? (LL-1 - tp) : tp;
    acc[((long)(b << 10) + tr) * DD + d] += outb[i] * (1.f + 0.1f * g);
}

__global__ void copyf(const float* __restrict__ src, float* __restrict__ dst, int n)
{
    int i = blockIdx.x * blockDim.x + threadIdx.x;
    if (i < n) dst[i] = src[i];
}

// ---------------------------------------------------------------------------
// Per-token LayerNorm over D=256 (one block per token)
// ---------------------------------------------------------------------------
__global__ void layernorm(const float* __restrict__ X, const float* __restrict__ g,
                          const float* __restrict__ bta, float* __restrict__ Y)
{
    __shared__ float red[DD];
    int tok = blockIdx.x, tid = threadIdx.x;
    float v = X[(long)tok * DD + tid];
    red[tid] = v; __syncthreads();
    for (int s = DD/2; s > 0; s >>= 1) { if (tid < s) red[tid] += red[tid + s]; __syncthreads(); }
    float mean = red[0] * (1.f / DD);
    __syncthreads();
    float c = v - mean;
    red[tid] = c * c; __syncthreads();
    for (int s = DD/2; s > 0; s >>= 1) { if (tid < s) red[tid] += red[tid + s]; __syncthreads(); }
    float var = red[0] * (1.f / DD);
    Y[(long)tok * DD + tid] = c * rsqrtf(var + 1e-5f) * g[tid] + bta[tid];
}

// ---------------------------------------------------------------------------
// Fused FFN (256->32 relu ->256) + residual + LayerNorm, one block per token
// ---------------------------------------------------------------------------
__global__ void ffn_ln(const float* __restrict__ Y3, const float* __restrict__ w1,
                       const float* __restrict__ b1, const float* __restrict__ w2,
                       const float* __restrict__ b2, const float* __restrict__ g,
                       const float* __restrict__ bta, float* __restrict__ Out)
{
    __shared__ float yl[DD];
    __shared__ float hl[UU];
    __shared__ float red[DD];
    int tok = blockIdx.x, tid = threadIdx.x;
    float y = Y3[(long)tok * DD + tid];
    yl[tid] = y;
    __syncthreads();
    if (tid < UU) {
        float a = b1[tid];
        for (int dd = 0; dd < DD; ++dd) a = fmaf(w1[tid*DD + dd], yl[dd], a);
        hl[tid] = fmaxf(a, 0.f);
    }
    __syncthreads();
    float o = b2[tid];
#pragma unroll
    for (int u = 0; u < UU; ++u) o = fmaf(w2[tid*UU + u], hl[u], o);
    float v = o + y;
    red[tid] = v; __syncthreads();
    for (int s = DD/2; s > 0; s >>= 1) { if (tid < s) red[tid] += red[tid + s]; __syncthreads(); }
    float mean = red[0] * (1.f / DD);
    __syncthreads();
    float c = v - mean;
    red[tid] = c * c; __syncthreads();
    for (int s = DD/2; s > 0; s >>= 1) { if (tid < s) red[tid] += red[tid + s]; __syncthreads(); }
    float var = red[0] * (1.f / DD);
    Out[(long)tok * DD + tid] = c * rsqrtf(var + 1e-5f) * g[tid] + bta[tid];
}

// ---------------------------------------------------------------------------
extern "C" void kernel_launch(void* const* d_in, const int* in_sizes, int n_in,
                              void* d_out, int out_size, void* d_ws, size_t ws_size,
                              hipStream_t stream)
{
    const float* x0 = (const float*)d_in[0];
    auto in = [&](int i){ return (const float*)d_in[i]; };

    float* ws   = (float*)d_ws;
    float* xbuf = ws;  ws += (long)MT*DD;        // round output / next input
    float* acc  = ws;  ws += (long)MT*DD;        // x + Y1 + Y2 accumulator
    float* y3   = ws;  ws += (long)MT*DD;        // LN1 output
    float* xz   = ws;  ws += (long)MT*2*DI;      // in_proj output (xm|res)
    float* xc   = ws;  ws += (long)MT*DI;        // conv/silu, then gated y
    float* dly  = ws;  ws += (long)MT*DI;        // delta, then scan y
    float* xdbl = ws;  ws += (long)MT*48;        // x_proj output (dt|B|C)
    float* outb = ws;  ws += (long)MT*DD;        // per-direction out_proj
    float* proj = ws;  ws += (long)MT*64;        // directional tanh proj
    float* Sv   = ws;  ws += 64;                 // rowsum of dir_pw
    float* dtwp = ws;  ws += (long)DI*32;        // zero-padded dt_w (K=32)

    const int TPB = 256;
    auto nblk = [](long n){ return (int)((n + 255) / 256); };

    for (int r = 0; r < 3; ++r) {
        const float* xin = (r == 0) ? x0 : xbuf;
        copyf<<<nblk((long)MT*DD), TPB, 0, stream>>>(xin, acc, MT*DD);

        for (int dir = 0; dir < 2; ++dir) {
            int base = (dir == 0) ? 1 : 14;
            const float* in_w = in(base+0)  + (long)r * 2*DI*DD;
            const float* cw   = in(base+1)  + (long)r * DI*KC;
            const float* cb   = in(base+2)  + (long)r * DI;
            const float* xpw  = in(base+3)  + (long)r * 48*DI;
            const float* dtw  = in(base+4)  + (long)r * DI*DTRK;
            const float* dtb  = in(base+5)  + (long)r * DI;
            const float* Alog = in(base+6)  + (long)r * DI*NS;
            const float* Dp   = in(base+7)  + (long)r * DI;
            const float* ow   = in(base+8)  + (long)r * DD*DI;
            const float* dpw  = in(base+9)  + (long)r * 64*DD;
            const float* dpb  = in(base+10) + (long)r * 64;
            const float* dgw  = in(base+11) + (long)r * DD*64;
            const float* dgb  = in(base+12) + (long)r * DD;
            int flip = dir;

            rowsum64<<<1, 64, 0, stream>>>(dpw, Sv);
            dir_proj<<<nblk((long)MT*64), TPB, 0, stream>>>(xin, Sv, dpb, proj, flip);
            pad_dtw<<<nblk(DI*32), TPB, 0, stream>>>(dtw, dtwp);

            // in_proj: xz = x(flip) @ in_w^T          M=8192 N=1024 K=256
            { int tiles = (MT/16)*(2*DI/16);
              gemm_bf16_wmma<DD><<<(tiles+7)/8, 256, 0, stream>>>(xin, DD, flip, in_w,
                                                                  xz, 2*DI, 2*DI, nullptr, 0); }
            conv_silu<<<nblk((long)MT*DI), TPB, 0, stream>>>(xz, cw, cb, xc);

            // x_proj: xdbl = xc @ xp_w^T              M=8192 N=48  K=512
            { int tiles = (MT/16)*(48/16);
              gemm_bf16_wmma<DI><<<(tiles+7)/8, 256, 0, stream>>>(xc, DI, 0, xpw,
                                                                  xdbl, 48, 48, nullptr, 0); }
            // delta = softplus(dt @ dtw_pad^T + dt_b) M=8192 N=512 K=32 (weight zero-padded)
            { int tiles = (MT/16)*(DI/16);
              gemm_bf16_wmma<32><<<(tiles+7)/8, 256, 0, stream>>>(xdbl, 48, 0, dtwp,
                                                                  dly, DI, DI, dtb, 1); }
            ssm_scan<<<BB*(DI/DCH), DCH, 0, stream>>>(xdbl, dly, xc, Alog);
            mix_gate_res<<<nblk((long)MT*DI), TPB, 0, stream>>>(xc, dly, xz, Dp);

            // out_proj: outb = y @ out_w^T            M=8192 N=256 K=512
            { int tiles = (MT/16)*(DD/16);
              gemm_bf16_wmma<DI><<<(tiles+7)/8, 256, 0, stream>>>(xc, DI, 0, ow,
                                                                  outb, DD, DD, nullptr, 0); }
            gate_accum<<<nblk((long)MT*DD), TPB, 0, stream>>>(outb, proj, dgw, dgb, acc, flip);
        }

        layernorm<<<MT, DD, 0, stream>>>(acc, in(27) + (long)r*DD, in(28) + (long)r*DD, y3);
        float* dst = (r == 2) ? (float*)d_out : xbuf;
        ffn_ln<<<MT, DD, 0, stream>>>(y3,
                                      in(31) + (long)r*UU*DD, in(32) + (long)r*UU,
                                      in(33) + (long)r*DD*UU, in(34) + (long)r*DD,
                                      in(29) + (long)r*DD,    in(30) + (long)r*DD, dst);
    }
}
